// CRF_644245094878
// MI455X (gfx1250) — matrix operand order repaired and checked by reading, not compiled, not run on hardware
//
#include <hip/hip_runtime.h>

// CRF loss for MI455X (gfx1250, wave32, WMMA).
// B=512, S=512, NUM_LABELS=64, L=66 (with start/end), SE_SCORE=-1000.
#define BB   512
#define SS   512
#define NL   64
#define LL   66
#define BT   16     // batch rows per wave (WMMA M)
#define LSTR 72     // LDS row stride in halves (64 K + 8 bank pad; 36 dwords)
#define L2E  1.44269504088896340736f
#define LN2  0.69314718055994530942f

typedef __attribute__((ext_vector_type(16))) _Float16 v16h;
typedef __attribute__((ext_vector_type(8)))  float    v8f;
typedef __attribute__((ext_vector_type(4)))  float    v4f;

union AFrag { v16h v; _Float16 e[16]; v4f q[2]; };  // 16x32 f16 A fragment
union CFrag { v8f  v; float    e[8];  };            // 16x16 f32 C/D fragment

__global__ void __launch_bounds__(1) crf_zero_out(float* out) { out[0] = 0.0f; }

// -------- gold path score: sum_b -(emissions + transitions)/B --------------
__global__ void __launch_bounds__(128)
crf_gold(const float* __restrict__ x, const int* __restrict__ y,
         const float* __restrict__ T, float* __restrict__ out) {
  const int b = blockIdx.x, tid = threadIdx.x;
  const int* yb = y + (size_t)b * SS;
  float p = 0.0f;
  for (int t = tid; t < SS; t += 128)      p += x[((size_t)b * SS + t) * NL + yb[t]];
  for (int t = tid; t < SS - 1; t += 128)  p += T[yb[t] * LL + yb[t + 1]];
  if (tid == 0) p += T[(LL - 2) * LL + yb[0]] + T[yb[SS - 1] * LL + (LL - 1)];
  p += __shfl_xor(p, 16); p += __shfl_xor(p, 8); p += __shfl_xor(p, 4);
  p += __shfl_xor(p, 2);  p += __shfl_xor(p, 1);
  __shared__ float red[4];
  if ((tid & 31) == 0) red[tid >> 5] = p;
  __syncthreads();
  if (tid == 0) atomicAdd(out, -(red[0] + red[1] + red[2] + red[3]) * (1.0f / (float)BB));
}

// -------- forward algorithm: linear-domain WMMA scan -----------------------
// One wave32 per 16-batch tile; alpha in 4 C-fragments (labels 0..63; labels
// 64/65 are exactly 0 in linear domain for t>=2).  E=exp(T) resident as 10
// f16 B-fragments.  Per step: exponent-only renorm (bit ops, no log/rcp),
// 2.25KB LDS transpose, 8 x v_wmma_f32_16x16x32_f16, multiply exp(emissions).
__global__ void __launch_bounds__(32)
crf_fwd(const float* __restrict__ x, const float* __restrict__ T,
        float* __restrict__ out) {
  __shared__ _Float16 vA[BT * LSTR];
  const int l     = threadIdx.x;   // 0..31
  const int half  = l >> 4;        // lane group (M split)
  const int lmod  = l & 15;
  const int bbase = blockIdx.x * BT;

  // Resident B fragments: E = exp(T), rows (K) 0..63, cols padded to 80.
  // B fragments, contiguous k order as A uses: lane l, element h -> K = h + 16*(l>>4), N = l%16
  v16h Bf[2][5];
  #pragma unroll
  for (int c = 0; c < 2; ++c) {
    #pragma unroll
    for (int n = 0; n < 5; ++n) {
      AFrag tmp;
      #pragma unroll
      for (int h = 0; h < 16; ++h) {
        const int i = 32 * c + h + 16 * half;   // prev label < 64
        const int j = 16 * n + lmod;            // next label
        float v = 0.0f;
        if (j < LL) v = __builtin_amdgcn_exp2f(T[i * LL + j] * L2E);
        tmp.e[h] = (_Float16)v;
      }
      Bf[c][n] = tmp.v;
    }
  }

  // ---- step t=1 directly: alpha_1[j] = exp(T[start,j] + x[b,0,j]) ----
  // C layout: element r -> M = r + 8*(l>>4), N = 16n + l%16
  float em[4][8];
  #pragma unroll
  for (int n = 0; n < 4; ++n)
    #pragma unroll
    for (int r = 0; r < 8; ++r)
      em[n][r] = x[((size_t)(bbase + r + 8 * half) * SS + 0) * NL + 16 * n + lmod];

  float t64[4];
  #pragma unroll
  for (int n = 0; n < 4; ++n) t64[n] = T[(LL - 2) * LL + 16 * n + lmod];

  CFrag acc[4];
  #pragma unroll
  for (int n = 0; n < 4; ++n)
    #pragma unroll
    for (int r = 0; r < 8; ++r)
      acc[n].e[r] = __builtin_amdgcn_exp2f((t64[n] + em[n][r]) * L2E);

  int esum[8];
  #pragma unroll
  for (int r = 0; r < 8; ++r) esum[r] = 0;

  // preload emissions for t=2 (x time index 1)
  #pragma unroll
  for (int n = 0; n < 4; ++n)
    #pragma unroll
    for (int r = 0; r < 8; ++r)
      em[n][r] = x[((size_t)(bbase + r + 8 * half) * SS + 1) * NL + 16 * n + lmod];

  #pragma unroll 1
  for (int t = 2; t <= SS; ++t) {
    // L2 prefetch ~6 steps ahead (2 x 128B lines cover a 256B row)
    const int tp = (t + 5 <= SS - 1) ? (t + 5) : (SS - 1);
    #pragma unroll
    for (int n = 0; n < 4; n += 2)
      #pragma unroll
      for (int r = 0; r < 8; ++r)
        __builtin_prefetch(&x[((size_t)(bbase + r + 8 * half) * SS + tp) * NL + 16 * n + lmod], 0, 3);

    // demand-load next step's emissions (double buffer)
    float emn[4][8];
    const int tn = (t <= SS - 1) ? t : (SS - 1);
    #pragma unroll
    for (int n = 0; n < 4; ++n)
      #pragma unroll
      for (int r = 0; r < 8; ++r)
        emn[n][r] = x[((size_t)(bbase + r + 8 * half) * SS + tn) * NL + 16 * n + lmod];

    // per-row max -> exact power-of-2 renorm (no log, no rcp; m is always
    // a normal float in [~2e-4, ~2e4])
    float inv[8];
    #pragma unroll
    for (int r = 0; r < 8; ++r) {
      float m = acc[0].e[r];
      #pragma unroll
      for (int n = 1; n < 4; ++n) m = fmaxf(m, acc[n].e[r]);
      m = fmaxf(m, __shfl_xor(m, 1));
      m = fmaxf(m, __shfl_xor(m, 2));
      m = fmaxf(m, __shfl_xor(m, 4));
      m = fmaxf(m, __shfl_xor(m, 8));
      const unsigned ef = __float_as_uint(m) >> 23;       // biased exponent
      esum[r] += (int)ef - 127;
      inv[r] = __uint_as_float((254u - ef) << 23);        // exact 2^-e
    }

    // stage normalized alpha (f16) into LDS [M][K]
    #pragma unroll
    for (int n = 0; n < 4; ++n)
      #pragma unroll
      for (int r = 0; r < 8; ++r)
        vA[(r + 8 * half) * LSTR + 16 * n + lmod] =
            (_Float16)(acc[n].e[r] * inv[r]);
    __builtin_amdgcn_wave_barrier();   // keep LDS store->load order

    // A fragments: M = l%16, K = h + 16*(l>>4)
    AFrag a[2];
    #pragma unroll
    for (int c = 0; c < 2; ++c) {
      const int base = lmod * LSTR + 32 * c + 16 * half;
      a[c].q[0] = *(const v4f*)(vA + base);
      a[c].q[1] = *(const v4f*)(vA + base + 8);
    }
    __builtin_amdgcn_wave_barrier();

    // 8 WMMAs: d[n] = sum_c A_c x E_cn
    CFrag d[4];
    #pragma unroll
    for (int n = 0; n < 4; ++n) {
      v8f z = {};
      #pragma unroll
      for (int c = 0; c < 2; ++c)
        z = __builtin_amdgcn_wmma_f32_16x16x32_f16(
                false, a[c].v, false, Bf[c][n], (short)0, z, false, false);
      d[n].v = z;
    }

    // apply emissions in linear domain
    #pragma unroll
    for (int n = 0; n < 4; ++n)
      #pragma unroll
      for (int r = 0; r < 8; ++r)
        acc[n].e[r] = d[n].e[r] * __builtin_amdgcn_exp2f(em[n][r] * L2E);

    #pragma unroll
    for (int n = 0; n < 4; ++n)
      #pragma unroll
      for (int r = 0; r < 8; ++r) em[n][r] = emn[n][r];
  }

  // ---- final step t=S+1: end row -> only column 65 (tile n=4) matters ----
  float inv[8];
  #pragma unroll
  for (int r = 0; r < 8; ++r) {
    float m = acc[0].e[r];
    #pragma unroll
    for (int n = 1; n < 4; ++n) m = fmaxf(m, acc[n].e[r]);
    m = fmaxf(m, __shfl_xor(m, 1));
    m = fmaxf(m, __shfl_xor(m, 2));
    m = fmaxf(m, __shfl_xor(m, 4));
    m = fmaxf(m, __shfl_xor(m, 8));
    const unsigned ef = __float_as_uint(m) >> 23;
    esum[r] += (int)ef - 127;
    inv[r] = __uint_as_float((254u - ef) << 23);
  }
  #pragma unroll
  for (int n = 0; n < 4; ++n)
    #pragma unroll
    for (int r = 0; r < 8; ++r)
      vA[(r + 8 * half) * LSTR + 16 * n + lmod] =
          (_Float16)(acc[n].e[r] * inv[r]);
  __builtin_amdgcn_wave_barrier();
  AFrag a[2];
  #pragma unroll
  for (int c = 0; c < 2; ++c) {
    const int base = lmod * LSTR + 32 * c + 16 * half;
    a[c].q[0] = *(const v4f*)(vA + base);
    a[c].q[1] = *(const v4f*)(vA + base + 8);
  }
  __builtin_amdgcn_wave_barrier();
  CFrag d4;
  {
    v8f z = {};
    #pragma unroll
    for (int c = 0; c < 2; ++c)
      z = __builtin_amdgcn_wmma_f32_16x16x32_f16(
              false, a[c].v, false, Bf[c][4], (short)0, z, false, false);
    d4.v = z;
  }

  // log_z[b] = log2(alpha_final[65]) * ln2 + esum * ln2
  float total = 0.0f;
  #pragma unroll
  for (int r = 0; r < 8; ++r) {
    float s = (lmod == 1) ? d4.e[r] : 0.0f;   // col 65 only
    s += __shfl_xor(s, 1); s += __shfl_xor(s, 2);
    s += __shfl_xor(s, 4); s += __shfl_xor(s, 8);
    total += (__builtin_amdgcn_logf(fmaxf(s, 1e-37f)) + (float)esum[r]) * LN2;
  }
  if (lmod == 0)  // lane 0 -> rows 0..7, lane 16 -> rows 8..15
    atomicAdd(out, total * (1.0f / (float)BB));
}

extern "C" void kernel_launch(void* const* d_in, const int* in_sizes, int n_in,
                              void* d_out, int out_size, void* d_ws, size_t ws_size,
                              hipStream_t stream) {
  (void)in_sizes; (void)n_in; (void)out_size; (void)d_ws; (void)ws_size;
  const float* x = (const float*)d_in[0];
  const int*   y = (const int*)d_in[1];
  const float* T = (const float*)d_in[2];
  float* out = (float*)d_out;

  crf_zero_out<<<1, 1, 0, stream>>>(out);
  crf_gold<<<BB, 128, 0, stream>>>(x, y, T, out);
  crf_fwd<<<BB / BT, 32, 0, stream>>>(x, T, out);
}
